// V1Attention_23639499997415
// MI455X (gfx1250) — matrix operand order repaired and checked by reading, compiled, not run on hardware
//
#include <hip/hip_runtime.h>
#include <hip/hip_bf16.h>

// MI455X (gfx1250) wave32 WMMA implementation of causal MHA.
// GEMMs via v_wmma_f32_16x16x32_f16 (f16 in, f32 accumulate), register
// double-buffered global->LDS staging; flash-attention K/V tiles via
// CDNA5 async load-to-LDS (ASYNCcnt) when the builtin is available.

typedef __attribute__((ext_vector_type(16))) _Float16 v16h;
typedef __attribute__((ext_vector_type(8)))  float    v8f;
typedef __attribute__((ext_vector_type(8)))  _Float16 h8;

constexpr int DIM      = 4096;
constexpr int N_HEADS  = 32;
constexpr int HEAD_DIM = 128;
constexpr int BATCH    = 2;
constexpr int SEQ      = 1024;
constexpr int ROWS     = BATCH * SEQ;          // 2048
constexpr float NEG_BIG = -1000000000.0f;      // matches reference mask value

#if defined(__gfx1250__) && __has_builtin(__builtin_amdgcn_global_load_async_to_lds_b128)
#define USE_ASYNC_LDS 1
#else
#define USE_ASYNC_LDS 0
#endif

#if USE_ASYNC_LDS
typedef int v4i __attribute__((vector_size(16)));
typedef __attribute__((address_space(1))) v4i* gv4i_p;   // global (AS1)
typedef __attribute__((address_space(3))) v4i* lv4i_p;   // LDS (AS3)

__device__ __forceinline__ void async_g2l_b128(const void* g, const void* lds) {
    gv4i_p gp = (gv4i_p)(uintptr_t)g;
    // generic LDS address low 32 bits == LDS offset (ISA 10.2 aperture rules)
    lv4i_p lp = (lv4i_p)(uint32_t)(uintptr_t)lds;
    __builtin_amdgcn_global_load_async_to_lds_b128(gp, lp, 0, 0);
}
__device__ __forceinline__ void wait_async0() {
#if __has_builtin(__builtin_amdgcn_s_wait_asynccnt)
    __builtin_amdgcn_s_wait_asynccnt(0);
#else
    asm volatile("s_wait_asynccnt 0x0" ::: "memory");
#endif
}
#endif

// ---------------------------------------------------------------------------
// Generic GEMM: out[M,N] = A[M,K] @ W[N,K]^T + bias[N]
// Block tile 128(M) x 128(N), K-step 32. 8 waves = 2(M) x 4(N); wave tile
// 64x32 = 4x2 accumulators of v_wmma_f32_16x16x32_f16.
// Global->LDS staging is register double-buffered: the next K-tile's global
// loads issue right after the barrier, so their waits hide behind the WMMAs.
// A may be f32 (converted to f16 while staging) or already f16.
// ---------------------------------------------------------------------------
template <typename AT, typename OT>
__global__ __launch_bounds__(256) void gemm_bias_wmma(
    const AT* __restrict__ A, const float* __restrict__ W,
    const float* __restrict__ bias, OT* __restrict__ out,
    int M, int N, int K)
{
    constexpr int BM = 128, BN = 128, BK = 32;
    __shared__ _Float16 As[BM][BK + 8];
    __shared__ _Float16 Ws[BN][BK + 8];

    const int tid  = threadIdx.x;
    const int lane = tid & 31;
    const int w    = tid >> 5;
    const int half = lane >> 4;
    const int l    = lane & 15;
    const int wm   = w & 1;          // wave M slot (0..1) -> 64 rows
    const int wn   = w >> 1;         // wave N slot (0..3) -> 32 cols
    const int row0 = blockIdx.y * BM;
    const int col0 = blockIdx.x * BN;

    // Per-thread staging: row r = tid/2, 16 contiguous k elements.
    const int r  = tid >> 1;
    const int kk = (tid & 1) * 16;
    const AT*    aSrc = A + (size_t)(row0 + r) * K + kk;
    const float* wSrc = W + (size_t)(col0 + r) * K + kk;

    AT    aReg[16];
    float wReg[16];
    #pragma unroll
    for (int i = 0; i < 16; ++i) { aReg[i] = aSrc[i]; wReg[i] = wSrc[i]; }

    v8f acc[4][2] = {};

    for (int k0 = 0; k0 < K; k0 += BK) {
        __syncthreads();   // previous compute done reading LDS (no-op 1st iter)
        #pragma unroll
        for (int i = 0; i < 16; ++i) {
            As[r][kk + i] = (_Float16)aReg[i];
            Ws[r][kk + i] = (_Float16)wReg[i];
        }
        __syncthreads();

        // Issue next tile's global loads now; waits resolve next iteration.
        if (k0 + BK < K) {
            #pragma unroll
            for (int i = 0; i < 16; ++i) {
                aReg[i] = aSrc[k0 + BK + i];
                wReg[i] = wSrc[k0 + BK + i];
            }
        }

        // A fragments (16x32 f16, contiguous k = half*16 + s per lane, same as B)
        v16h a[4], b[2];
        #pragma unroll
        for (int mi = 0; mi < 4; ++mi) {
            const int m = wm * 64 + mi * 16 + l;
            #pragma unroll
            for (int v = 0; v < 8; ++v) {
                const int kc = half * 16 + v * 2;
                a[mi][2 * v]     = As[m][kc];
                a[mi][2 * v + 1] = As[m][kc + 1];
            }
        }
        // B fragments (32x16): lane holds column, k = half*16 + 2v
        #pragma unroll
        for (int ni = 0; ni < 2; ++ni) {
            const int n = wn * 32 + ni * 16 + l;
            #pragma unroll
            for (int v = 0; v < 8; ++v) {
                const int kc = half * 16 + v * 2;
                b[ni][2 * v]     = Ws[n][kc];
                b[ni][2 * v + 1] = Ws[n][kc + 1];
            }
        }
        #pragma unroll
        for (int mi = 0; mi < 4; ++mi)
            #pragma unroll
            for (int ni = 0; ni < 2; ++ni)
                acc[mi][ni] = __builtin_amdgcn_wmma_f32_16x16x32_f16(
                    false, a[mi], false, b[ni], (short)0, acc[mi][ni], false, false);
    }

    // Epilogue: C layout row = half*8 + r, col = l
    #pragma unroll
    for (int mi = 0; mi < 4; ++mi)
        #pragma unroll
        for (int ni = 0; ni < 2; ++ni)
            #pragma unroll
            for (int rr = 0; rr < 8; ++rr) {
                const int m = row0 + wm * 64 + mi * 16 + half * 8 + rr;
                const int n = col0 + wn * 32 + ni * 16 + l;
                out[(size_t)m * N + n] = (OT)(acc[mi][ni][rr] + bias[n]);
            }
}

// ---------------------------------------------------------------------------
// RoPE (interleaved pairs, fp32 math) + transpose [B,S,H*hd] -> [B,H,S,hd]
// ---------------------------------------------------------------------------
__global__ __launch_bounds__(64) void rope_transpose(
    const _Float16* __restrict__ q, const _Float16* __restrict__ k,
    const _Float16* __restrict__ v,
    _Float16* __restrict__ qh, _Float16* __restrict__ kh,
    _Float16* __restrict__ vh)
{
    const int h = blockIdx.x, s = blockIdx.y, b = blockIdx.z;
    const int d = threadIdx.x;                 // pair index 0..63
    const size_t src = ((size_t)(b * SEQ + s)) * DIM + h * HEAD_DIM + 2 * d;
    const size_t dst = (((size_t)(b * N_HEADS + h)) * SEQ + s) * HEAD_DIM + 2 * d;

    const float inv_freq = __powf(10000.0f, -(float)d / 64.0f);
    float sn, cs;
    __sincosf((float)s * inv_freq, &sn, &cs);

    const float q0 = (float)q[src], q1 = (float)q[src + 1];
    qh[dst]     = (_Float16)(q0 * cs - q1 * sn);
    qh[dst + 1] = (_Float16)(q0 * sn + q1 * cs);
    const float k0 = (float)k[src], k1 = (float)k[src + 1];
    kh[dst]     = (_Float16)(k0 * cs - k1 * sn);
    kh[dst + 1] = (_Float16)(k0 * sn + k1 * cs);
    vh[dst]     = v[src];
    vh[dst + 1] = v[src + 1];
}

// ---------------------------------------------------------------------------
// Flash attention: one block = 128 q rows of one (b,h); 8 waves x 16 rows.
// KV processed in shared 64-key tiles (async copied to LDS on CDNA5).
// Causal mask applied analytically (same -1e9 value as the reference mask).
// Output written as f16 in [B,S,H*hd] layout, ready for the Wo GEMM.
// ---------------------------------------------------------------------------
__global__ __launch_bounds__(256) void flash_attn(
    const _Float16* __restrict__ qh, const _Float16* __restrict__ kh,
    const _Float16* __restrict__ vh, _Float16* __restrict__ attn)
{
    constexpr int QT = 128;   // q rows per block
    constexpr int KT = 64;    // keys per iteration
    __shared__ _Float16 Ks[KT][HEAD_DIM + 8];
    __shared__ _Float16 Vs[KT][HEAD_DIM + 8];
    __shared__ _Float16 Ps[8][16][KT + 8];   // per-wave P scratch (C -> A relayout)

    const int tid  = threadIdx.x;
    const int lane = tid & 31;
    const int w    = tid >> 5;
    const int half = lane >> 4;
    const int l    = lane & 15;
    const int qBase = blockIdx.x * QT;
    const int h = blockIdx.y, b = blockIdx.z;
    const size_t bh = (size_t)(b * N_HEADS + h) * SEQ;
    const float inv_sqrt = 0.08838834764831845f;   // 1/sqrt(128)
    const int qRow = qBase + w * 16;

    // Q fragments for the wave's 16 rows: 4 chunks over hd=128 (held in VGPRs)
    v16h aq[4];
    {
        const _Float16* qp = qh + (bh + qRow + l) * HEAD_DIM;
        #pragma unroll
        for (int c = 0; c < 4; ++c)
            #pragma unroll
            for (int v = 0; v < 8; ++v) {
                const int kk = c * 32 + half * 16 + v * 2;
                aq[c][2 * v]     = qp[kk];
                aq[c][2 * v + 1] = qp[kk + 1];
            }
    }

    v8f o[8] = {};
    float run_max[8], run_sum[8];
    #pragma unroll
    for (int r = 0; r < 8; ++r) { run_max[r] = -3.0e38f; run_sum[r] = 0.0f; }

    const int kvEnd = qBase + QT;   // causal: later keys fully masked for block
    for (int kv0 = 0; kv0 < kvEnd; kv0 += KT) {
        // Cooperative K/V tile staging: 64 x 128 halves each, 16B granules
        #pragma unroll
        for (int i = 0; i < 4; ++i) {
            const int chunk = tid + i * 256;
            const int r  = chunk >> 4;
            const int cc = (chunk & 15) * 8;
            const size_t g = (bh + kv0 + r) * HEAD_DIM + cc;
#if USE_ASYNC_LDS
            async_g2l_b128(kh + g, &Ks[r][cc]);
            async_g2l_b128(vh + g, &Vs[r][cc]);
#else
            *(h8*)&Ks[r][cc] = *(const h8*)(kh + g);
            *(h8*)&Vs[r][cc] = *(const h8*)(vh + g);
#endif
        }
#if USE_ASYNC_LDS
        wait_async0();
#endif
        __syncthreads();

        // Scores S = Q K^T : 4 n-tiles x 4 k-chunks = 16 WMMAs
        v8f sc[4] = {};
        #pragma unroll
        for (int jt = 0; jt < 4; ++jt)
            #pragma unroll
            for (int c = 0; c < 4; ++c) {
                v16h bk;
                #pragma unroll
                for (int v = 0; v < 8; ++v) {
                    const int kk = c * 32 + half * 16 + v * 2;
                    bk[2 * v]     = Ks[jt * 16 + l][kk];
                    bk[2 * v + 1] = Ks[jt * 16 + l][kk + 1];
                }
                sc[jt] = __builtin_amdgcn_wmma_f32_16x16x32_f16(
                    false, aq[c], false, bk, (short)0, sc[jt], false, false);
            }

        // Online softmax (row = half*8 + r lives across the 16 lanes of a half)
        float alpha[8];
        #pragma unroll
        for (int r = 0; r < 8; ++r) {
            const int qpos = qRow + half * 8 + r;
            float mx = -3.0e38f;
            #pragma unroll
            for (int jt = 0; jt < 4; ++jt) {
                const int kvpos = kv0 + jt * 16 + l;
                float sv = sc[jt][r] * inv_sqrt + (kvpos <= qpos ? 0.0f : NEG_BIG);
                sc[jt][r] = sv;
                mx = fmaxf(mx, sv);
            }
            #pragma unroll
            for (int m = 1; m < 16; m <<= 1)
                mx = fmaxf(mx, __shfl_xor(mx, m, 32));
            const float nm = fmaxf(run_max[r], mx);
            alpha[r] = __expf(run_max[r] - nm);
            float ps = 0.0f;
            #pragma unroll
            for (int jt = 0; jt < 4; ++jt) {
                const float p = __expf(sc[jt][r] - nm);
                Ps[w][half * 8 + r][jt * 16 + l] = (_Float16)p;
                ps += p;
            }
            #pragma unroll
            for (int m = 1; m < 16; m <<= 1)
                ps += __shfl_xor(ps, m, 32);
            run_sum[r] = run_sum[r] * alpha[r] + ps;
            run_max[r] = nm;
        }
        #pragma unroll
        for (int nt = 0; nt < 8; ++nt)
            #pragma unroll
            for (int r = 0; r < 8; ++r)
                o[nt][r] *= alpha[r];

        // O += P V : relayout P via per-wave LDS scratch into A fragments
        v16h ap[2];
        #pragma unroll
        for (int c = 0; c < 2; ++c)
            #pragma unroll
            for (int v = 0; v < 8; ++v) {
                const int kk = c * 32 + half * 16 + v * 2;
                ap[c][2 * v]     = Ps[w][l][kk];
                ap[c][2 * v + 1] = Ps[w][l][kk + 1];
            }
        #pragma unroll
        for (int nt = 0; nt < 8; ++nt)
            #pragma unroll
            for (int c = 0; c < 2; ++c) {
                v16h bv;
                #pragma unroll
                for (int v = 0; v < 8; ++v) {
                    const int kk = c * 32 + half * 16 + v * 2;
                    bv[2 * v]     = Vs[kk][nt * 16 + l];
                    bv[2 * v + 1] = Vs[kk + 1][nt * 16 + l];
                }
                o[nt] = __builtin_amdgcn_wmma_f32_16x16x32_f16(
                    false, ap[c], false, bv, (short)0, o[nt], false, false);
            }
        __syncthreads();
    }

    // Normalize and write [B,S,H*hd] f16 (input to Wo GEMM)
    #pragma unroll
    for (int nt = 0; nt < 8; ++nt)
        #pragma unroll
        for (int r = 0; r < 8; ++r) {
            const int qpos = qRow + half * 8 + r;
            const float ov = o[nt][r] / run_sum[r];
            attn[((size_t)(b * SEQ + qpos)) * DIM + h * HEAD_DIM + nt * 16 + l] =
                (_Float16)ov;
        }
}

// ---------------------------------------------------------------------------
extern "C" void kernel_launch(void* const* d_in, const int* in_sizes, int n_in,
                              void* d_out, int out_size, void* d_ws, size_t ws_size,
                              hipStream_t stream)
{
    (void)in_sizes; (void)n_in; (void)out_size; (void)ws_size;
    const float* x  = (const float*)d_in[0];
    const float* wq = (const float*)d_in[1];
    const float* bq = (const float*)d_in[2];
    const float* wk = (const float*)d_in[3];
    const float* bk = (const float*)d_in[4];
    const float* wv = (const float*)d_in[5];
    const float* bv = (const float*)d_in[6];
    const float* wo = (const float*)d_in[7];
    const float* bo = (const float*)d_in[8];
    // d_in[9] = mask (causal, applied analytically), d_in[10] = start_pos (0)
    float* out = (float*)d_out;

    char* ws = (char*)d_ws;
    const size_t SZ = (size_t)ROWS * DIM * sizeof(_Float16);   // 16 MB
    _Float16* q16 = (_Float16*)(ws + 0 * SZ);
    _Float16* k16 = (_Float16*)(ws + 1 * SZ);
    _Float16* v16 = (_Float16*)(ws + 2 * SZ);
    _Float16* qh  = (_Float16*)(ws + 3 * SZ);
    _Float16* kh  = (_Float16*)(ws + 4 * SZ);
    _Float16* vh  = (_Float16*)(ws + 5 * SZ);
    _Float16* attn16 = q16;   // q16 dead after rope; reuse for attention output

    const dim3 gg(DIM / 128, ROWS / 128), gb(256);
    gemm_bias_wmma<float, _Float16><<<gg, gb, 0, stream>>>(x, wq, bq, q16, ROWS, DIM, DIM);
    gemm_bias_wmma<float, _Float16><<<gg, gb, 0, stream>>>(x, wk, bk, k16, ROWS, DIM, DIM);
    gemm_bias_wmma<float, _Float16><<<gg, gb, 0, stream>>>(x, wv, bv, v16, ROWS, DIM, DIM);

    rope_transpose<<<dim3(N_HEADS, SEQ, BATCH), 64, 0, stream>>>(q16, k16, v16, qh, kh, vh);

    flash_attn<<<dim3(SEQ / 128, N_HEADS, BATCH), 256, 0, stream>>>(qh, kh, vh, attn16);

    gemm_bias_wmma<_Float16, float><<<gg, gb, 0, stream>>>(attn16, wo, bo, out, ROWS, DIM, DIM);
}
